// GATv3_22411139350786
// MI455X (gfx1250) — compile-verified, matrix operand order checked by reading, not run on hardware
//
#include <hip/hip_runtime.h>
#include <hip/hip_bf16.h>
#include <math.h>

typedef __attribute__((ext_vector_type(16))) __bf16        v16bf;
typedef __attribute__((ext_vector_type(8)))  float         v8f;
typedef __attribute__((ext_vector_type(8)))  unsigned int  v8u;

#define NEGC (-9.0e15f)
#define LDSTRIDE 20   // 16 data dwords + 4 pad: conflict-free & 16B-aligned fragment groups

// two f32 -> one packed-bf16 dword (RNE) in a single VALU instruction
__device__ __forceinline__ unsigned int pack2_bf16(float x, float y) {
  unsigned int r;
  asm("v_cvt_pk_bf16_f32 %0, %1, %2" : "=v"(r) : "v"(x), "v"(y));
  return r;
}

// C[M,N] = A[M,K] @ op(B)  (+bias[n])  (optional mask epilogue: adj>0 ? acc : NEG)
// transB==0: B is [K,N] row-major.  transB==1: B is [N,K] row-major (Q @ K^T).
// M,N multiples of 128; K multiple of 32.
__global__ void __launch_bounds__(256)
gemm_bf16_wmma(const float* __restrict__ A,
               const float* __restrict__ B,
               const float* __restrict__ bias,
               const float* __restrict__ adjmask,
               float* __restrict__ C,
               int M, int N, int K, int transB)
{
  __shared__ unsigned int ldsA[2][128 * LDSTRIDE];   // double-buffered
  __shared__ unsigned int ldsB[2][128 * LDSTRIDE];

  const int tid   = threadIdx.x;
  const int lane  = tid & 31;
  const int wave  = tid >> 5;
  const int wm    = wave >> 1;            // 0..3  (M direction)
  const int wn    = wave & 1;             // 0..1  (N direction)
  const int mBase = blockIdx.y * 128;
  const int nBase = blockIdx.x * 128;

  const v8f vzero = {0.f,0.f,0.f,0.f,0.f,0.f,0.f,0.f};
  v8f acc[2][4];
  #pragma unroll
  for (int i = 0; i < 2; ++i)
    #pragma unroll
    for (int j = 0; j < 4; ++j) acc[i][j] = vzero;

  const int laneRow = lane & 15;
  const int khalf   = lane >> 4;          // which K-half of the fragment this lane holds

  int buf = 0;
  for (int k0 = 0; k0 < K; k0 += 32, buf ^= 1) {
    // ======== STAGE k0 into ldsX[buf] (overlaps previous iteration's WMMA drain) ========
    // -- issue ALL global loads first so they form one clause with incremental waits --
    float4 va[4];
    #pragma unroll
    for (int j = 0; j < 4; ++j) {
      int q  = tid + 256 * j;             // 0..1023 float4 index
      int m  = q >> 3;                    // 0..127 (8 float4 per 32-float row)
      int qc = q & 7;                     // float4 within row -> pairs 2qc, 2qc+1
      va[j] = *(const float4*)(A + (size_t)(mBase + m) * K + k0 + 4 * qc);
    }
    float4 vb0[2], vb1[2], vbt[4];
    if (transB == 0) {
      #pragma unroll
      for (int j = 0; j < 2; ++j) {
        int q  = tid + 256 * j;           // (kd, n-quad)
        int kd = q >> 5;                  // 0..15
        int n4 = q & 31;                  // n = 4*n4 .. 4*n4+3
        const float* bp = B + (size_t)(k0 + 2 * kd) * N + nBase + 4 * n4;
        vb0[j] = *(const float4*)(bp);
        vb1[j] = *(const float4*)(bp + N);
      }
    } else {
      #pragma unroll
      for (int j = 0; j < 4; ++j) {
        int q  = tid + 256 * j;
        int n  = q >> 3;
        int qc = q & 7;
        vbt[j] = *(const float4*)(B + (size_t)(nBase + n) * K + k0 + 4 * qc);
      }
    }
    // prefetch two K-steps ahead (emits global_prefetch_b8)
    if (k0 + 64 < K)
      __builtin_prefetch(A + (size_t)(mBase + (tid >> 1)) * K + (k0 + 64), 0, 1);

    // -- convert + LDS stores --
    #pragma unroll
    for (int j = 0; j < 4; ++j) {
      int q  = tid + 256 * j;
      int m  = q >> 3;
      int qc = q & 7;
      ldsA[buf][m * LDSTRIDE + 2 * qc]     = pack2_bf16(va[j].x, va[j].y);
      ldsA[buf][m * LDSTRIDE + 2 * qc + 1] = pack2_bf16(va[j].z, va[j].w);
    }
    if (transB == 0) {
      #pragma unroll
      for (int j = 0; j < 2; ++j) {
        int q  = tid + 256 * j;
        int kd = q >> 5;
        int nb = (q & 31) * 4;
        ldsB[buf][(nb + 0) * LDSTRIDE + kd] = pack2_bf16(vb0[j].x, vb1[j].x);
        ldsB[buf][(nb + 1) * LDSTRIDE + kd] = pack2_bf16(vb0[j].y, vb1[j].y);
        ldsB[buf][(nb + 2) * LDSTRIDE + kd] = pack2_bf16(vb0[j].z, vb1[j].z);
        ldsB[buf][(nb + 3) * LDSTRIDE + kd] = pack2_bf16(vb0[j].w, vb1[j].w);
      }
    } else {
      #pragma unroll
      for (int j = 0; j < 4; ++j) {
        int q  = tid + 256 * j;
        int n  = q >> 3;
        int qc = q & 7;
        ldsB[buf][n * LDSTRIDE + 2 * qc]     = pack2_bf16(vbt[j].x, vbt[j].y);
        ldsB[buf][n * LDSTRIDE + 2 * qc + 1] = pack2_bf16(vbt[j].z, vbt[j].w);
      }
    }
    __syncthreads();   // single barrier per K-step (covers this stage + prior compute)

    // ======== COMPUTE from ldsX[buf] ========
    v8u af[2], bfr[4];
    #pragma unroll
    for (int mt = 0; mt < 2; ++mt) {
      int row = wm * 32 + mt * 16 + laneRow;
      unsigned b = row * LDSTRIDE + khalf * 4;
      #pragma unroll
      for (int i = 0; i < 4; ++i) { af[mt][i] = ldsA[buf][b + i]; af[mt][4 + i] = ldsA[buf][b + 8 + i]; }
    }
    #pragma unroll
    for (int nt = 0; nt < 4; ++nt) {
      int col = wn * 64 + nt * 16 + laneRow;
      unsigned b = col * LDSTRIDE + khalf * 4;
      #pragma unroll
      for (int i = 0; i < 4; ++i) { bfr[nt][i] = ldsB[buf][b + i]; bfr[nt][4 + i] = ldsB[buf][b + 8 + i]; }
    }
    #pragma unroll
    for (int mt = 0; mt < 2; ++mt)
      #pragma unroll
      for (int nt = 0; nt < 4; ++nt)
        acc[mt][nt] = __builtin_amdgcn_wmma_f32_16x16x32_bf16(
            false, __builtin_bit_cast(v16bf, af[mt]),
            false, __builtin_bit_cast(v16bf, bfr[nt]),
            (short)0, acc[mt][nt], false, false);
    // no trailing barrier: next stage writes the other LDS buffer
  }

  // ---- epilogue: C/D layout lanes0-15: M=r, lanes16-31: M=8+r ----
  const int rowOff = (lane >> 4) * 8;
  const int cn     = lane & 15;
  #pragma unroll
  for (int mt = 0; mt < 2; ++mt) {
    #pragma unroll
    for (int nt = 0; nt < 4; ++nt) {
      const int n  = nBase + wn * 64 + nt * 16 + cn;
      const float bv = bias ? bias[n] : 0.f;
      #pragma unroll
      for (int r = 0; r < 8; ++r) {
        const int m = mBase + wm * 32 + mt * 16 + rowOff + r;
        float v = acc[mt][nt][r] + bv;
        if (adjmask) v = (adjmask[(size_t)m * N + n] > 0.f) ? v : NEGC;
        C[(size_t)m * N + n] = v;
      }
    }
  }
}

// in-place row softmax over [4096, 4096]; one block per row; row register-resident
__global__ void __launch_bounds__(256)
softmax_rows_4096(float* __restrict__ S)
{
  __shared__ float red[256];
  float* p = S + (size_t)blockIdx.x * 4096;
  const int tid = threadIdx.x;
  float x[16];
  float mx = -3.0e38f;
  #pragma unroll
  for (int j = 0; j < 16; ++j) { x[j] = p[tid + 256 * j]; mx = fmaxf(mx, x[j]); }
  red[tid] = mx; __syncthreads();
  for (int s = 128; s > 0; s >>= 1) { if (tid < s) red[tid] = fmaxf(red[tid], red[tid + s]); __syncthreads(); }
  mx = red[0]; __syncthreads();
  float sum = 0.f;
  #pragma unroll
  for (int j = 0; j < 16; ++j) { x[j] = __expf(x[j] - mx); sum += x[j]; }
  red[tid] = sum; __syncthreads();
  for (int s = 128; s > 0; s >>= 1) { if (tid < s) red[tid] += red[tid + s]; __syncthreads(); }
  const float inv = 1.f / red[0];
  #pragma unroll
  for (int j = 0; j < 16; ++j) p[tid + 256 * j] = x[j] * inv;
}

// out-row (len 512) = mask[idx] * row / max(||row||2, 1e-12), written into concat buffer
__global__ void __launch_bounds__(256)
l2norm_scale(const float* __restrict__ T, const float* __restrict__ maskp, int maskIdx,
             float* __restrict__ F, int colOff)
{
  __shared__ float red[256];
  const float* p = T + (size_t)blockIdx.x * 512;
  const int tid = threadIdx.x;
  const float a = p[tid], b = p[tid + 256];
  red[tid] = a * a + b * b; __syncthreads();
  for (int s = 128; s > 0; s >>= 1) { if (tid < s) red[tid] += red[tid + s]; __syncthreads(); }
  const float scale = maskp[maskIdx] / fmaxf(sqrtf(red[0]), 1e-12f);
  float* out = F + (size_t)blockIdx.x * 1536 + colOff;
  out[tid] = a * scale;
  out[tid + 256] = b * scale;
}

// logits = relu(F) @ Wc + bc, then log_softmax over 40 classes; one block per row
__global__ void __launch_bounds__(64)
classifier_logsoftmax(const float* __restrict__ F, const float* __restrict__ Wc,
                      const float* __restrict__ bc, float* __restrict__ out)
{
  __shared__ float vals[40];
  __shared__ float stats[2];
  const int row = blockIdx.x, tid = threadIdx.x;
  const float* f = F + (size_t)row * 1536;
  if (tid < 40) {
    float acc = bc[tid];
    for (int k = 0; k < 1536; ++k) {
      float v = f[k];
      v = v > 0.f ? v : 0.f;
      acc += v * Wc[(size_t)k * 40 + tid];
    }
    vals[tid] = acc;
  }
  __syncthreads();
  if (tid == 0) {
    float mx = vals[0];
    for (int c = 1; c < 40; ++c) mx = fmaxf(mx, vals[c]);
    float s = 0.f;
    for (int c = 0; c < 40; ++c) s += __expf(vals[c] - mx);
    stats[0] = mx; stats[1] = __logf(s);
  }
  __syncthreads();
  if (tid < 40) out[(size_t)row * 40 + tid] = vals[tid] - stats[0] - stats[1];
}

__global__ void hop_softmax(const float* __restrict__ hs, float* __restrict__ mask)
{
  if (threadIdx.x == 0 && blockIdx.x == 0) {
    const float a = hs[0], b = hs[1], c = hs[2];
    const float mx = fmaxf(a, fmaxf(b, c));
    const float ea = __expf(a - mx), eb = __expf(b - mx), ec = __expf(c - mx);
    const float inv = 1.f / (ea + eb + ec);
    mask[0] = ea * inv; mask[1] = eb * inv; mask[2] = ec * inv;
  }
}

extern "C" void kernel_launch(void* const* d_in, const int* in_sizes, int n_in,
                              void* d_out, int out_size, void* d_ws, size_t ws_size,
                              hipStream_t stream) {
  (void)in_sizes; (void)n_in; (void)out_size; (void)ws_size;
  const float* x      = (const float*)d_in[0];
  const float* adjatt = (const float*)d_in[1];
  const float* adj1   = (const float*)d_in[2];
  const float* adj2   = (const float*)d_in[3];
  const float* hs     = (const float*)d_in[4];
  const float* Wfc[3] = {(const float*)d_in[5], (const float*)d_in[7], (const float*)d_in[9]};
  const float* bfc[3] = {(const float*)d_in[6], (const float*)d_in[8], (const float*)d_in[10]};
  const float* Wq[2]  = {(const float*)d_in[11], (const float*)d_in[15]};
  const float* bq[2]  = {(const float*)d_in[12], (const float*)d_in[16]};
  const float* Wk[2]  = {(const float*)d_in[13], (const float*)d_in[17]};
  const float* bk[2]  = {(const float*)d_in[14], (const float*)d_in[18]};
  const float* Wcls   = (const float*)d_in[19];
  const float* bcls   = (const float*)d_in[20];

  // workspace layout (floats)
  float* ws   = (float*)d_ws;
  float* mask = ws;                         // 64
  float* T    = ws + 64;                    // [4096,512] current tmp
  float* TA   = T  + 2097152;               // t_att
  float* QW   = TA + 2097152;               // Q/K pre-propagation scratch
  float* Qb   = QW + 2097152;               // Q
  float* Kb   = Qb + 2097152;               // K
  float* TB   = Kb + 2097152;               // att @ tmp
  float* Fc   = TB + 2097152;               // [4096,1536] concat buffer
  float* Sb   = Fc + 6291456;               // [4096,4096] layer-1 scores/att

  float* logits = (float*)d_out;            // [4096,40]
  float* att2   = logits + 4096 * 40;       // [4096,4096] == tmp_1 output

  const dim3 blk(256);
  const dim3 gN512(512 / 128, 4096 / 128);  // GEMMs with N=512
  const dim3 gN4096(4096 / 128, 4096 / 128);

  hop_softmax<<<1, 32, 0, stream>>>(hs, mask);

  // ---- layer 0: tmp = x@W0 + b0 ; normalize into Fc[:,0:512] ----
  gemm_bf16_wmma<<<gN512, blk, 0, stream>>>(x, Wfc[0], bfc[0], nullptr, T, 4096, 512, 512, 0);
  l2norm_scale<<<4096, 256, 0, stream>>>(T, mask, 0, Fc, 0);

  // ---- layer 1 (attention over adj1) ----
  gemm_bf16_wmma<<<gN512, blk, 0, stream>>>(x, Wfc[1], bfc[1], nullptr, T, 4096, 512, 512, 0);
  gemm_bf16_wmma<<<gN512, blk, 0, stream>>>(adjatt, T,  nullptr, nullptr, TA, 4096, 512, 4096, 0);
  gemm_bf16_wmma<<<gN512, blk, 0, stream>>>(TA, Wq[0], bq[0], nullptr, QW, 4096, 512, 512, 0);
  gemm_bf16_wmma<<<gN512, blk, 0, stream>>>(adjatt, QW, nullptr, nullptr, Qb, 4096, 512, 4096, 0);
  gemm_bf16_wmma<<<gN512, blk, 0, stream>>>(TA, Wk[0], bk[0], nullptr, QW, 4096, 512, 512, 0);
  gemm_bf16_wmma<<<gN512, blk, 0, stream>>>(adjatt, QW, nullptr, nullptr, Kb, 4096, 512, 4096, 0);
  gemm_bf16_wmma<<<gN4096, blk, 0, stream>>>(Qb, Kb, nullptr, adj1, Sb, 4096, 4096, 512, 1);
  softmax_rows_4096<<<4096, 256, 0, stream>>>(Sb);
  gemm_bf16_wmma<<<gN512, blk, 0, stream>>>(Sb, T, nullptr, nullptr, TB, 4096, 512, 4096, 0);
  l2norm_scale<<<4096, 256, 0, stream>>>(TB, mask, 1, Fc, 512);

  // ---- layer 2 (attention over adj2); att written straight into d_out (== tmp_1) ----
  gemm_bf16_wmma<<<gN512, blk, 0, stream>>>(x, Wfc[2], bfc[2], nullptr, T, 4096, 512, 512, 0);
  gemm_bf16_wmma<<<gN512, blk, 0, stream>>>(adjatt, T,  nullptr, nullptr, TA, 4096, 512, 4096, 0);
  gemm_bf16_wmma<<<gN512, blk, 0, stream>>>(TA, Wq[1], bq[1], nullptr, QW, 4096, 512, 512, 0);
  gemm_bf16_wmma<<<gN512, blk, 0, stream>>>(adjatt, QW, nullptr, nullptr, Qb, 4096, 512, 4096, 0);
  gemm_bf16_wmma<<<gN512, blk, 0, stream>>>(TA, Wk[1], bk[1], nullptr, QW, 4096, 512, 512, 0);
  gemm_bf16_wmma<<<gN512, blk, 0, stream>>>(adjatt, QW, nullptr, nullptr, Kb, 4096, 512, 4096, 0);
  gemm_bf16_wmma<<<gN4096, blk, 0, stream>>>(Qb, Kb, nullptr, adj2, att2, 4096, 4096, 512, 1);
  softmax_rows_4096<<<4096, 256, 0, stream>>>(att2);
  gemm_bf16_wmma<<<gN512, blk, 0, stream>>>(att2, T, nullptr, nullptr, TB, 4096, 512, 4096, 0);
  l2norm_scale<<<4096, 256, 0, stream>>>(TB, mask, 2, Fc, 1024);

  // ---- classifier + log_softmax ----
  classifier_logsoftmax<<<4096, 64, 0, stream>>>(Fc, Wcls, bcls, logits);
}
